// GaussianBackbone3D_1417339207912
// MI455X (gfx1250) — matrix operand order repaired and checked by reading, not compiled, not run on hardware
//
#include <hip/hip_runtime.h>
#include <hip/hip_bf16.h>
#include <stdint.h>
#include <math.h>

#define GXD 704
#define GYD 200
#define GZD 32
#define NVOX 200000
#define CINF 64
#define CHID 128
#define KSEL 10000

// output layout (floats)
static constexpr size_t O_MU = 0;
static constexpr size_t O_S  = 30000;
static constexpr size_t O_R  = 60000;
static constexpr size_t O_Q  = 100000;
static constexpr size_t O_E  = 740000;
static constexpr size_t O_XY = 780000;          // 64*140800
static constexpr size_t O_XZ = 9791200;         // 64*22528
static constexpr size_t O_YZ = 11232992;        // 64*6400
static constexpr int PXY = GYD * GXD;           // 140800
static constexpr int PXZ = GXD * GZD;           // 22528
static constexpr int PYZ = GYD * GZD;           // 6400

typedef __attribute__((ext_vector_type(16))) __bf16 v16bf;
typedef __attribute__((ext_vector_type(8)))  float  v8f;

union ABFrag {
    v16bf    v;
    uint16_t u[16];
    uint4    q[2];
};

__device__ __forceinline__ uint16_t f2bf(float f) {
    uint32_t u = __float_as_uint(f);
    uint32_t r = u + 0x7FFFu + ((u >> 16) & 1u);   // round-to-nearest-even
    return (uint16_t)(r >> 16);
}

// ---------------- utility fills ----------------
__global__ void fill_i32(int* p, size_t n, int v) {
    for (size_t i = blockIdx.x * (size_t)blockDim.x + threadIdx.x; i < n;
         i += (size_t)gridDim.x * blockDim.x) p[i] = v;
}
__global__ void fill_f32(float* p, size_t n, float v) {
    for (size_t i = blockIdx.x * (size_t)blockDim.x + threadIdx.x; i < n;
         i += (size_t)gridDim.x * blockDim.x) p[i] = v;
}

// ---------------- grid map ----------------
__global__ void grid_scatter(const int* __restrict__ coords, int* __restrict__ grid) {
    int i = blockIdx.x * blockDim.x + threadIdx.x;
    if (i >= NVOX) return;
    int z = coords[(size_t)i * 4 + 1];
    int y = coords[(size_t)i * 4 + 2];
    int x = coords[(size_t)i * 4 + 3];
    grid[((size_t)z * GYD + y) * GXD + x] = i;
}

// ---------------- f32 -> bf16 cast ----------------
__global__ void cast_bf16(const float* __restrict__ in, uint16_t* __restrict__ out, size_t n) {
    for (size_t i = blockIdx.x * (size_t)blockDim.x + threadIdx.x; i < n;
         i += (size_t)gridDim.x * blockDim.x) out[i] = f2bf(in[i]);
}

// ---------------- weight pack into WMMA-B lane layout ----------------
// packed[tap][kc][tile][lane][j] ; k = kc*32 + (lane>=16?16:0) + j ; n = tile*16 + (lane&15)
template <int CIN_T>
__global__ void pack_weights(const float* __restrict__ w, uint16_t* __restrict__ wp) {
    constexpr int KC = CIN_T / 32;
    const int total = 27 * KC * 8 * 32 * 16;
    for (int i = blockIdx.x * blockDim.x + threadIdx.x; i < total;
         i += gridDim.x * blockDim.x) {
        int j    = i & 15;
        int lane = (i >> 4) & 31;
        int tile = (i >> 9) & 7;
        int kc   = (i >> 12) % KC;
        int tap  = (i >> 12) / KC;
        int k = kc * 32 + ((lane >> 4) << 4) + j;
        int n = tile * 16 + (lane & 15);
        wp[i] = f2bf(w[((size_t)tap * CIN_T + k) * CHID + n]);
    }
}

// ---------------- submanifold conv via WMMA bf16 ----------------
// Block = 32 voxels x 128 out channels. 8 waves, each wave owns one 16-col N tile
// and two 16-row M tiles (two accumulators) -> each B fragment feeds 2 WMMAs.
template <int CIN_T, bool OUT_BF16>
__global__ __launch_bounds__(256) void subm_conv_wmma(
    const uint16_t* __restrict__ feats, const uint16_t* __restrict__ wpack,
    const int* __restrict__ coords, const int* __restrict__ grid,
    const float* __restrict__ gamma, const float* __restrict__ beta,
    uint16_t* __restrict__ out_bf, float* __restrict__ out_f) {
    constexpr int KC = CIN_T / 32;
    constexpr int CH = CIN_T / 8;   // uint4 chunks per feature row
    __shared__ __align__(16) uint16_t atile[32][CIN_T];
    __shared__ int nid[27][32];
    __shared__ int cz[32], cy[32], cx[32];

    const int t    = threadIdx.x;
    const int vox0 = blockIdx.x * 32;

    if (t < 32) {
        const int* c4 = coords + (size_t)(vox0 + t) * 4;
        cz[t] = c4[1]; cy[t] = c4[2]; cx[t] = c4[3];
    }
    __syncthreads();
    for (int e = t; e < 27 * 32; e += 256) {
        int tap = e >> 5, m = e & 31;
        int dz = tap / 9 - 1, dy = (tap / 3) % 3 - 1, dx = tap % 3 - 1;
        int nz = cz[m] + dz, ny = cy[m] + dy, nx = cx[m] + dx;
        int idx = -1;
        if (nz >= 0 && nz < GZD && ny >= 0 && ny < GYD && nx >= 0 && nx < GXD)
            idx = grid[((size_t)nz * GYD + ny) * GXD + nx];
        nid[tap][m] = idx;
    }

    const int lane = t & 31;
    const int wave = t >> 5;        // out-channel tile 0..7
    const int half = lane >> 4;
    const int mrow = lane & 15;

    v8f acc0 = {0.f, 0.f, 0.f, 0.f, 0.f, 0.f, 0.f, 0.f};
    v8f acc1 = {0.f, 0.f, 0.f, 0.f, 0.f, 0.f, 0.f, 0.f};

    for (int tap = 0; tap < 27; ++tap) {
        __syncthreads();            // protect atile reuse (and nid on first iter)
        for (int e = t; e < 32 * CH; e += 256) {
            int row = e / CH, ch = e % CH;
            int src = nid[tap][row];
            uint4 val = make_uint4(0u, 0u, 0u, 0u);
            if (src >= 0)
                val = *(const uint4*)(feats + (size_t)src * CIN_T + ch * 8);
            *(uint4*)(&atile[row][ch * 8]) = val;
        }
        __syncthreads();
        // prefetch next tap's weights for this wave/lane (global_prefetch_b8)
        if (tap + 1 < 27) {
            const uint16_t* np =
                wpack + ((((size_t)(tap + 1) * KC) * 8 + wave) * 32 + lane) * 16;
            __builtin_prefetch(np, 0, 1);
        }
#pragma unroll
        for (int kc = 0; kc < KC; ++kc) {
            ABFrag a0, a1, b;
            const uint16_t* bp =
                wpack + ((((size_t)tap * KC + kc) * 8 + wave) * 32 + lane) * 16;
            b.q[0] = *(const uint4*)(bp);
            b.q[1] = *(const uint4*)(bp + 8);
            // A fragment uses contiguous k order chosen here, not the ISA table order (matches packed B): lanes 0-15 hold K 0..15,
            // lanes 16-31 hold K 16..31
            const uint16_t* ar0 = &atile[mrow][kc * 32 + half * 16];
            a0.q[0] = *(const uint4*)(ar0);
            a0.q[1] = *(const uint4*)(ar0 + 8);
            const uint16_t* ar1 = &atile[16 + mrow][kc * 32 + half * 16];
            a1.q[0] = *(const uint4*)(ar1);
            a1.q[1] = *(const uint4*)(ar1 + 8);
            acc0 = __builtin_amdgcn_wmma_f32_16x16x32_bf16(
                false, a0.v, false, b.v, (short)0, acc0, false, false);
            acc1 = __builtin_amdgcn_wmma_f32_16x16x32_bf16(
                false, a1.v, false, b.v, (short)0, acc1, false, false);
        }
    }

    // fused BN(eval) + ReLU epilogue; C/D layout: lane half picks M block, VGPR r = row offset
    const int col = wave * 16 + mrow;
    const float sc = gamma[col] * 0.9999950000374997f;   // 1/sqrt(1+1e-5)
    const float sh = beta[col];
#pragma unroll
    for (int r = 0; r < 8; ++r) {
        int row = half * 8 + r;
        float v0 = fmaxf(acc0[r] * sc + sh, 0.f);
        float v1 = fmaxf(acc1[r] * sc + sh, 0.f);
        size_t o0 = (size_t)(vox0 + row) * CHID + col;
        size_t o1 = (size_t)(vox0 + 16 + row) * CHID + col;
        if (OUT_BF16) { out_bf[o0] = f2bf(v0); out_bf[o1] = f2bf(v1); }
        else          { out_f[o0]  = v0;       out_f[o1]  = v1; }
    }
}

// ---------------- semantic head: logits -> softmax -> score histogram ----------------
__global__ void sem_head(const float* __restrict__ h2, const float* __restrict__ sw,
                         const float* __restrict__ sb, float* __restrict__ probs,
                         int* __restrict__ bins, int* __restrict__ hist) {
    int i = blockIdx.x * blockDim.x + threadIdx.x;
    if (i >= NVOX) return;
    float l0 = sb[0], l1 = sb[1], l2 = sb[2], l3 = sb[3];
    const float* hr = h2 + (size_t)i * CHID;
    for (int c = 0; c < CHID; ++c) {
        float v = hr[c];
        l0 += v * sw[c * 4 + 0];
        l1 += v * sw[c * 4 + 1];
        l2 += v * sw[c * 4 + 2];
        l3 += v * sw[c * 4 + 3];
    }
    float mx = fmaxf(fmaxf(l0, l1), fmaxf(l2, l3));
    float e0 = __expf(l0 - mx), e1 = __expf(l1 - mx);
    float e2 = __expf(l2 - mx), e3 = __expf(l3 - mx);
    float inv = 1.f / (e0 + e1 + e2 + e3);
    float p0 = e0 * inv, p1 = e1 * inv, p2 = e2 * inv, p3 = e3 * inv;
    probs[(size_t)i * 4 + 0] = p0; probs[(size_t)i * 4 + 1] = p1;
    probs[(size_t)i * 4 + 2] = p2; probs[(size_t)i * 4 + 3] = p3;
    int pred = 0; float best = p0;
    if (p1 > best) { best = p1; pred = 1; }
    if (p2 > best) { best = p2; pred = 2; }
    if (p3 > best) { best = p3; pred = 3; }
    if (pred != 0) {
        float fg = fmaxf(fmaxf(p1, p2), p3);
        int b = min(255, (int)(fg * 256.f));
        bins[i] = b;
        atomicAdd(hist + b, 1);
    } else {
        bins[i] = -1;
    }
}

__global__ void find_thresh(const int* __restrict__ hist, int* __restrict__ thr) {
    if (threadIdx.x == 0 && blockIdx.x == 0) {
        int cum = 0, T = 0;
        for (int b = 255; b >= 0; --b) {
            cum += hist[b];
            if (cum >= KSEL) { T = b; break; }
        }
        thr[0] = T;
    }
}

__global__ void select_topk(const int* __restrict__ bins, const int* __restrict__ thr,
                            int* __restrict__ sel, int* __restrict__ cnt) {
    int i = blockIdx.x * blockDim.x + threadIdx.x;
    if (i >= NVOX) return;
    int b = bins[i];
    if (b < 0 || b < thr[0]) return;
    int pos = atomicAdd(cnt, 1);
    if (pos < KSEL) sel[pos] = i;
}

// ---------------- per-selected-row heads ----------------
__global__ __launch_bounds__(128) void heads_kernel(
    const float* __restrict__ h2, const int* __restrict__ sel,
    const float* __restrict__ pw, const float* __restrict__ pb,
    const float* __restrict__ sbias, const int* __restrict__ coords,
    const float* __restrict__ probs,
    const float* __restrict__ m1w, const float* __restrict__ m1b,
    const float* __restrict__ m2w, const float* __restrict__ m2b,
    const float* __restrict__ m3w, const float* __restrict__ m3b,
    float* __restrict__ out) {
    __shared__ float hrow[128];
    __shared__ float pr[71];
    __shared__ float rn;
    const int row = blockIdx.x;
    const int vi  = sel[row];
    const int t   = threadIdx.x;
    hrow[t] = h2[(size_t)vi * CHID + t];
    __syncthreads();
    if (t < 71) {
        float s = pb[t];
        for (int c = 0; c < 128; ++c) s += hrow[c] * pw[c * 71 + t];
        pr[t] = s;
    }
    __syncthreads();
    if (t == 0) {
        float v0 = pr[3] + 1e-6f, v1 = pr[4] + 1e-6f;
        float v2 = pr[5] + 1e-6f, v3 = pr[6] + 1e-6f;
        rn = fmaxf(sqrtf(v0 * v0 + v1 * v1 + v2 * v2 + v3 * v3), 1e-12f);
    }
    __syncthreads();
    if (t < 3) {
        float x = pr[t] + sbias[t];
        float sp = (x > 20.f) ? x : log1pf(__expf(x));
        out[O_S + (size_t)row * 3 + t] = fminf(fmaxf(sp, 0.05f), 1.5f);
        const int* c4 = coords + (size_t)vi * 4;
        float m;
        if (t == 0)      m = (float)c4[3] * 0.1f + 0.0f;
        else if (t == 1) m = (float)c4[2] * 0.4f - 40.0f;
        else             m = (float)c4[1] * 0.125f - 2.0f;
        out[O_MU + (size_t)row * 3 + t] = m;
    }
    if (t < 4)  out[O_R + (size_t)row * 4 + t] = (pr[3 + t] + 1e-6f) / rn;
    if (t < 64) out[O_Q + (size_t)row * 64 + t] = pr[7 + t];
    if (t == 0) {
        float p0 = probs[(size_t)vi * 4 + 0], p1 = probs[(size_t)vi * 4 + 1];
        float p2 = probs[(size_t)vi * 4 + 2], p3 = probs[(size_t)vi * 4 + 3];
        float e1a[8], e2a[8];
#pragma unroll
        for (int j = 0; j < 8; ++j) {
            float s = m1b[j] + p0 * m1w[0 * 8 + j] + p1 * m1w[1 * 8 + j] +
                      p2 * m1w[2 * 8 + j] + p3 * m1w[3 * 8 + j];
            e1a[j] = fmaxf(s, 0.f);
        }
#pragma unroll
        for (int j = 0; j < 8; ++j) {
            float s = m2b[j];
#pragma unroll
            for (int i2 = 0; i2 < 8; ++i2) s += e1a[i2] * m2w[i2 * 8 + j];
            e2a[j] = fmaxf(s, 0.f);
        }
#pragma unroll
        for (int j = 0; j < 4; ++j) {
            float s = m3b[j];
#pragma unroll
            for (int i2 = 0; i2 < 8; ++i2) s += e2a[i2] * m3w[i2 * 4 + j];
            out[O_E + (size_t)row * 4 + j] = s;
        }
    }
}

// ---------------- TPV scatter-mean ----------------
__global__ void tpv_scatter(const float* __restrict__ vf, const int* __restrict__ coords,
                            float* __restrict__ out, float* __restrict__ cxy,
                            float* __restrict__ cxz, float* __restrict__ cyz) {
    const size_t total = (size_t)NVOX * 64;
    for (size_t i = blockIdx.x * (size_t)blockDim.x + threadIdx.x; i < total;
         i += (size_t)gridDim.x * blockDim.x) {
        int vi = (int)(i >> 6), c = (int)(i & 63);
        float f = vf[i];
        int z = coords[(size_t)vi * 4 + 1];
        int y = coords[(size_t)vi * 4 + 2];
        int x = coords[(size_t)vi * 4 + 3];
        atomicAdd(out + O_XY + (size_t)c * PXY + (size_t)y * GXD + x, f);
        atomicAdd(out + O_XZ + (size_t)c * PXZ + (size_t)x * GZD + z, f);
        atomicAdd(out + O_YZ + (size_t)c * PYZ + (size_t)y * GZD + z, f);
        if (c == 0) {
            atomicAdd(cxy + (size_t)y * GXD + x, 1.f);
            atomicAdd(cxz + (size_t)x * GZD + z, 1.f);
            atomicAdd(cyz + (size_t)y * GZD + z, 1.f);
        }
    }
}

__global__ void plane_norm(float* __restrict__ plane, const float* __restrict__ cnt, int P) {
    int total = 64 * P;
    for (int i = blockIdx.x * blockDim.x + threadIdx.x; i < total;
         i += gridDim.x * blockDim.x) {
        int p = i % P;
        plane[i] = plane[i] / fmaxf(cnt[p], 1.f);
    }
}

// ---------------- launcher ----------------
extern "C" void kernel_launch(void* const* d_in, const int* in_sizes, int n_in,
                              void* d_out, int out_size, void* d_ws, size_t ws_size,
                              hipStream_t stream) {
    (void)in_sizes; (void)n_in; (void)out_size; (void)ws_size;
    const float* vf    = (const float*)d_in[0];
    const float* w1    = (const float*)d_in[1];
    const float* bn1g  = (const float*)d_in[2];
    const float* bn1b  = (const float*)d_in[3];
    const float* w2    = (const float*)d_in[4];
    const float* bn2g  = (const float*)d_in[5];
    const float* bn2b  = (const float*)d_in[6];
    const float* semw  = (const float*)d_in[7];
    const float* semb  = (const float*)d_in[8];
    const float* pw    = (const float*)d_in[9];
    const float* pb    = (const float*)d_in[10];
    const float* sbias = (const float*)d_in[11];
    const float* m1w   = (const float*)d_in[12];
    const float* m1b   = (const float*)d_in[13];
    const float* m2w   = (const float*)d_in[14];
    const float* m2b   = (const float*)d_in[15];
    const float* m3w   = (const float*)d_in[16];
    const float* m3b   = (const float*)d_in[17];
    const int*   coords = (const int*)d_in[18];
    float* out = (float*)d_out;

    char*  ws  = (char*)d_ws;
    size_t off = 0;
    auto alloc = [&](size_t bytes) -> char* {
        char* p = ws + off;
        off += (bytes + 255) & ~(size_t)255;
        return p;
    };
    int*      grid = (int*)     alloc((size_t)GZD * GYD * GXD * 4);
    uint16_t* fbf  = (uint16_t*)alloc((size_t)NVOX * CINF * 2);
    uint16_t* w1p  = (uint16_t*)alloc((size_t)27 * CINF * CHID * 2);
    uint16_t* w2p  = (uint16_t*)alloc((size_t)27 * CHID * CHID * 2);
    uint16_t* h1   = (uint16_t*)alloc((size_t)NVOX * CHID * 2);
    float*    h2   = (float*)   alloc((size_t)NVOX * CHID * 4);
    float*    probs= (float*)   alloc((size_t)NVOX * 4 * 4);
    int*      bins = (int*)     alloc((size_t)NVOX * 4);
    int*      hist = (int*)     alloc(256 * 4);
    int*      thr  = (int*)     alloc(4);
    int*      sel  = (int*)     alloc((size_t)KSEL * 4);
    int*      selc = (int*)     alloc(4);
    float*    cxy  = (float*)   alloc((size_t)PXY * 4);
    float*    cxz  = (float*)   alloc((size_t)PXZ * 4);
    float*    cyz  = (float*)   alloc((size_t)PYZ * 4);

    // ---- init ----
    fill_i32<<<4096, 256, 0, stream>>>(grid, (size_t)GZD * GYD * GXD, -1);
    fill_i32<<<64, 256, 0, stream>>>(hist, 256, 0);
    fill_i32<<<64, 256, 0, stream>>>(thr, 1, 0);
    fill_i32<<<64, 256, 0, stream>>>(sel, KSEL, 0);
    fill_i32<<<1, 32, 0, stream>>>(selc, 1, 0);
    fill_f32<<<1024, 256, 0, stream>>>(cxy, (size_t)(PXY + PXZ + PYZ), 0.f); // contiguous
    fill_f32<<<8192, 256, 0, stream>>>(out + O_XY,
                                       (size_t)64 * (PXY + PXZ + PYZ), 0.f);

    // ---- preprocessing ----
    grid_scatter<<<(NVOX + 255) / 256, 256, 0, stream>>>(coords, grid);
    cast_bf16<<<8192, 256, 0, stream>>>(vf, fbf, (size_t)NVOX * CINF);
    pack_weights<CINF><<<(27 * 2 * 8 * 32 * 16 + 255) / 256, 256, 0, stream>>>(w1, w1p);
    pack_weights<CHID><<<(27 * 4 * 8 * 32 * 16 + 255) / 256, 256, 0, stream>>>(w2, w2p);

    // ---- WMMA submanifold convs (fused BN+ReLU), 32 voxels per block ----
    subm_conv_wmma<CINF, true><<<NVOX / 32, 256, 0, stream>>>(
        fbf, w1p, coords, grid, bn1g, bn1b, h1, nullptr);
    subm_conv_wmma<CHID, false><<<NVOX / 32, 256, 0, stream>>>(
        h1, w2p, coords, grid, bn2g, bn2b, nullptr, h2);

    // ---- semantic head + top-K ----
    sem_head<<<(NVOX + 255) / 256, 256, 0, stream>>>(h2, semw, semb, probs, bins, hist);
    find_thresh<<<1, 32, 0, stream>>>(hist, thr);
    select_topk<<<(NVOX + 255) / 256, 256, 0, stream>>>(bins, thr, sel, selc);

    // ---- per-selection heads ----
    heads_kernel<<<KSEL, 128, 0, stream>>>(h2, sel, pw, pb, sbias, coords, probs,
                                           m1w, m1b, m2w, m2b, m3w, m3b, out);

    // ---- TPV scatter-mean ----
    tpv_scatter<<<8192, 256, 0, stream>>>(vf, coords, out, cxy, cxz, cyz);
    plane_norm<<<(64 * PXY + 255) / 256, 256, 0, stream>>>(out + O_XY, cxy, PXY);
    plane_norm<<<(64 * PXZ + 255) / 256, 256, 0, stream>>>(out + O_XZ, cxz, PXZ);
    plane_norm<<<(64 * PYZ + 255) / 256, 256, 0, stream>>>(out + O_YZ, cyz, PYZ);
}
